// PosSelfAtten_64604898066773
// MI455X (gfx1250) — compile-verified, matrix operand order checked by reading, not run on hardware
//
#include <hip/hip_runtime.h>
#include <hip/hip_bf16.h>

#define HW   4096
#define NB   8
#define CIN  256
#define CQ   128
#define CV   256

typedef float    v2f  __attribute__((ext_vector_type(2)));
typedef float    v8f  __attribute__((ext_vector_type(8)));
typedef _Float16 v16h __attribute__((ext_vector_type(16)));
typedef _Float16 h8   __attribute__((ext_vector_type(8)));

// Load one f16 A/B-matrix fragment (16x32 K-chunk) for a wave32 WMMA.
// Per the CDNA5 ISA layout, lane's half h maps to K = (h&7) + 8*(lane>>4) + 16*(h>>3),
// so with row-major (row = M for A, row = N for B) data each lane needs exactly two
// contiguous 16-byte loads from its row.
static __device__ __forceinline__ v16h load_frag16(const _Float16* rowBase, int lh) {
  const h8 lo = *reinterpret_cast<const h8*>(rowBase + 8 * lh);
  const h8 hi = *reinterpret_cast<const h8*>(rowBase + 8 * lh + 16);
  v16h r;
#pragma unroll
  for (int i = 0; i < 8; ++i) { r[i] = lo[i]; r[i + 8] = hi[i]; }
  return r;
}

// ---------------------------------------------------------------------------
// Projection: D[o,p] = sum_c W[o,c] * x[n,c,p], then BN(eval)+ReLU, cast f16.
// Uses exact fp32 WMMA (v_wmma_f32_16x16x4_f32). One wave = one 16x16 tile.
// posMajor==0: store feat as fT[n][o][p]  (V operand layout for attention)
// posMajor==1: store q    as q[n][p][o]   (Q/K operand layout for attention)
// ---------------------------------------------------------------------------
__global__ __launch_bounds__(128) void proj_bn_relu_kernel(
    const float* __restrict__ x, const float* __restrict__ w,
    const float* __restrict__ gamma, const float* __restrict__ beta,
    const float* __restrict__ mean, const float* __restrict__ var,
    _Float16* __restrict__ outh, int posMajor)
{
  const int lane = threadIdx.x;
  const int l15  = lane & 15;
  const int lh   = lane >> 4;
  const int p0   = (blockIdx.x * blockDim.y + threadIdx.y) * 16;
  const int o0   = blockIdx.y * 16;
  const int n    = blockIdx.z;

  const float* wrow = w + (size_t)(o0 + l15) * CIN;              // A: M = l15
  const float* xcol = x + (size_t)n * CIN * HW + (p0 + l15);     // B: N = l15

  v8f acc0 = {};
  v8f acc1 = {};
#pragma unroll 4
  for (int c0 = 0; c0 < CIN; c0 += 8) {
    // fp32 A frag: K = 2*lh + v ; fp32 B frag: K = 2*lh + v (mirrored layout)
    const float2 aw0 = *reinterpret_cast<const float2*>(wrow + c0 + 2 * lh);
    const float2 aw1 = *reinterpret_cast<const float2*>(wrow + c0 + 4 + 2 * lh);
    v2f a0; a0[0] = aw0.x; a0[1] = aw0.y;
    v2f a1; a1[0] = aw1.x; a1[1] = aw1.y;
    v2f b0; b0[0] = xcol[(size_t)(c0 + 2 * lh + 0) * HW];
            b0[1] = xcol[(size_t)(c0 + 2 * lh + 1) * HW];
    v2f b1; b1[0] = xcol[(size_t)(c0 + 4 + 2 * lh + 0) * HW];
            b1[1] = xcol[(size_t)(c0 + 4 + 2 * lh + 1) * HW];
    acc0 = __builtin_amdgcn_wmma_f32_16x16x4_f32(false, a0, false, b0, (short)0, acc0, false, false);
    acc1 = __builtin_amdgcn_wmma_f32_16x16x4_f32(false, a1, false, b1, (short)0, acc1, false, false);
  }
  const v8f acc = acc0 + acc1;

  // C/D layout: VGPR r of this lane holds element (M = r + 8*lh, N = l15).
#pragma unroll
  for (int r = 0; r < 8; ++r) {
    const int ch   = o0 + 8 * lh + r;
    const float iv = gamma[ch] * rsqrtf(var[ch] + 1e-5f);
    const float bb = beta[ch] - mean[ch] * iv;
    float v = acc[r] * iv + bb;
    v = v > 0.f ? v : 0.f;
    const _Float16 hv = (_Float16)v;
    const int p = p0 + l15;
    if (posMajor)
      outh[((size_t)n * HW + p) * CQ + ch] = hv;     // q[n][p][c]
    else
      outh[((size_t)n * CV + ch) * HW + p] = hv;     // fT[n][c][p]
  }
}

// ---------------------------------------------------------------------------
// Flash attention: per wave, one 16-row tile i0..i0+15.
//   S(16x32) = Q(16x128) . K^T  via 8x v_wmma_f32_16x16x32_f16
//   online softmax in base-2 (energies pre-scaled by log2(e); exact identity)
//   O(16x256) += P(16x32) . V(32x256) via 16x WMMA  (P transposed through LDS)
// ---------------------------------------------------------------------------
__global__ __launch_bounds__(128) void flash_attn_kernel(
    const _Float16* __restrict__ q, const _Float16* __restrict__ fT,
    const float* __restrict__ mask, float* __restrict__ out)
{
  // 1/(1e-8 + sqrt(128)) folded with log2(e): softmax(x) == softmax_base2(x*log2e)
  constexpr float LOG2E     = 1.4426950408889634f;
  constexpr float ATT_SCALE = (1.0f / (1e-8f + 11.313708498984761f)) * LOG2E;
  __shared__ __align__(16) _Float16 Pbuf[4][16 * 32];

  const int lane = threadIdx.x;
  const int l15  = lane & 15;
  const int lh   = lane >> 4;
  const int wid  = threadIdx.y;
  const int i0   = (blockIdx.x * 4 + wid) * 16;
  const int n    = blockIdx.y;

  // Resident Q A-fragments (16 rows x 128 K, four K=32 chunks)
  v16h qa[4];
  {
    const _Float16* qrow = q + ((size_t)n * HW + i0 + l15) * CQ;
#pragma unroll
    for (int kc = 0; kc < 4; ++kc) qa[kc] = load_frag16(qrow + kc * 32, lh);
  }

  v8f oacc[16];
#pragma unroll
  for (int t = 0; t < 16; ++t) oacc[t] = (v8f){};
  float mrow[8], lrow[8];
#pragma unroll
  for (int r = 0; r < 8; ++r) { mrow[r] = -1e30f; lrow[r] = 0.f; }

  _Float16* pb    = &Pbuf[wid][0];
  const float* mk = mask + (size_t)n * HW;

  for (int j0 = 0; j0 < HW; j0 += 32) {
    // Prefetch next block's K rows toward the WGP (global_prefetch_b8);
    // working set is L2-resident, this hides L2->L0 latency behind the WMMAs.
    if (j0 + 32 < HW) {
      __builtin_prefetch(q + ((size_t)n * HW + j0 + 32 + l15) * CQ, 0, 3);
      __builtin_prefetch(q + ((size_t)n * HW + j0 + 48 + l15) * CQ, 0, 3);
    }

    // ---- S = Q . K^T (K rows j0..j0+31 of q serve as B fragments) ----
    v8f s0 = (v8f){};
    v8f s1 = (v8f){};
#pragma unroll
    for (int kc = 0; kc < 4; ++kc) {
      const _Float16* kb = q + ((size_t)n * HW + j0 + l15) * CQ + kc * 32;
      const v16h b0 = load_frag16(kb, lh);
      const v16h b1 = load_frag16(kb + 16 * CQ, lh);
      s0 = __builtin_amdgcn_wmma_f32_16x16x32_f16(false, qa[kc], false, b0, (short)0, s0, false, false);
      s1 = __builtin_amdgcn_wmma_f32_16x16x32_f16(false, qa[kc], false, b1, (short)0, s1, false, false);
    }

    // additive mask term depends only on column j (also in base-2 domain)
    const float ma0 = (mk[j0 + l15]      - 1.f) * (1e8f * LOG2E);
    const float ma1 = (mk[j0 + 16 + l15] - 1.f) * (1e8f * LOG2E);

    // ---- online softmax over this 32-column block (base-2) ----
    v8f avec;
#pragma unroll
    for (int r = 0; r < 8; ++r) {
      float e0 = s0[r] * ATT_SCALE + ma0;
      float e1 = s1[r] * ATT_SCALE + ma1;
      float vmax = fmaxf(e0, e1);
#pragma unroll
      for (int m = 1; m <= 8; m <<= 1) vmax = fmaxf(vmax, __shfl_xor(vmax, m, 32));
      const float mn = fmaxf(mrow[r], vmax);
      avec[r] = exp2f(mrow[r] - mn);
      mrow[r] = mn;
      e0 = exp2f(e0 - mn);
      e1 = exp2f(e1 - mn);
      float rs = e0 + e1;
#pragma unroll
      for (int m = 1; m <= 8; m <<= 1) rs += __shfl_xor(rs, m, 32);
      lrow[r] = lrow[r] * avec[r] + rs;
      s0[r] = e0;
      s1[r] = e1;
    }
    // rescale running O with packed f32 math (v_pk_mul_f32)
#pragma unroll
    for (int t = 0; t < 16; ++t) oacc[t] *= avec;

    // ---- P: C/D layout -> A layout via this wave's private LDS tile ----
#pragma unroll
    for (int r = 0; r < 8; ++r) {
      pb[(8 * lh + r) * 32 + l15]      = (_Float16)s0[r];
      pb[(8 * lh + r) * 32 + 16 + l15] = (_Float16)s1[r];
    }
    asm volatile("s_wait_dscnt 0" ::: "memory");   // same-wave LDS RAW
    const v16h pa = load_frag16(pb + l15 * 32, lh);

    // ---- O += P . V over all 256 output channels ----
#pragma unroll
    for (int t = 0; t < 16; ++t) {
      const _Float16* vrow = fT + ((size_t)n * CV + t * 16 + l15) * HW + j0;
      const v16h vb = load_frag16(vrow, lh);
      oacc[t] = __builtin_amdgcn_wmma_f32_16x16x32_f16(false, pa, false, vb, (short)0, oacc[t], false, false);
    }
  }

  // ---- normalize, apply output mask, store fp32 [n][c][i] ----
  float fin[8];
#pragma unroll
  for (int r = 0; r < 8; ++r)
    fin[r] = (1.0f / lrow[r]) * mk[i0 + 8 * lh + r];

#pragma unroll
  for (int t = 0; t < 16; ++t) {
    float* dst = out + ((size_t)n * CV + t * 16 + l15) * HW + i0 + 8 * lh;
    float4 lo4 = make_float4(oacc[t][0] * fin[0], oacc[t][1] * fin[1],
                             oacc[t][2] * fin[2], oacc[t][3] * fin[3]);
    float4 hi4 = make_float4(oacc[t][4] * fin[4], oacc[t][5] * fin[5],
                             oacc[t][6] * fin[6], oacc[t][7] * fin[7]);
    *reinterpret_cast<float4*>(dst)     = lo4;
    *reinterpret_cast<float4*>(dst + 4) = hi4;
  }
}

extern "C" void kernel_launch(void* const* d_in, const int* in_sizes, int n_in,
                              void* d_out, int out_size, void* d_ws, size_t ws_size,
                              hipStream_t stream) {
  (void)in_sizes; (void)n_in; (void)out_size; (void)ws_size;
  const float* x  = (const float*)d_in[0];
  const float* mk = (const float*)d_in[1];
  const float* rw = (const float*)d_in[2];
  const float* rg = (const float*)d_in[3];
  const float* rb = (const float*)d_in[4];
  const float* rm = (const float*)d_in[5];
  const float* rv = (const float*)d_in[6];
  const float* aw = (const float*)d_in[7];
  const float* ag = (const float*)d_in[8];
  const float* ab = (const float*)d_in[9];
  const float* am = (const float*)d_in[10];
  const float* av = (const float*)d_in[11];
  float* out = (float*)d_out;

  _Float16* qbuf = (_Float16*)d_ws;                      // [8][4096][128] f16 (8 MB)
  _Float16* fT   = qbuf + (size_t)NB * HW * CQ;          // [8][256][4096] f16 (16 MB)

  const dim3 blk(32, 4);
  proj_bn_relu_kernel<<<dim3(HW / 64, CV / 16, NB), blk, 0, stream>>>(
      x, rw, rg, rb, rm, rv, fT, 0);
  proj_bn_relu_kernel<<<dim3(HW / 64, CQ / 16, NB), blk, 0, stream>>>(
      x, aw, ag, ab, am, av, qbuf, 1);
  flash_attn_kernel<<<dim3(HW / 64, NB), blk, 0, stream>>>(qbuf, fT, mk, out);
}
